// RetrievalSoftmax_70849780515086
// MI455X (gfx1250) — matrix operand order repaired and checked by reading, not compiled, not run on hardware
//
#include <hip/hip_runtime.h>
#include <hip/hip_fp16.h>

typedef __attribute__((ext_vector_type(16))) _Float16 v16h;
typedef __attribute__((ext_vector_type(8)))  _Float16 v8h;
typedef __attribute__((ext_vector_type(8)))  float    v8f;

#define B_Q   1024
#define N_DB  100000
#define N_PAD 100032        // 1563 * 64: exact cover for 64-wide n tiles
#define DIM   256
#define TOPK  70
#define INV_TEMP 25.0f      // 1/0.04

// ---------------------------------------------------------------------------
// Kernel 1: row L2-normalize fp32 -> fp16 (one row per 256-thread block).
// Rows >= `rows` (padding) are written as zeros without touching src.
// ---------------------------------------------------------------------------
__global__ void k_normalize_f16(const float* __restrict__ src,
                                _Float16* __restrict__ dst, int rows) {
  const int row = blockIdx.x;
  const int t   = threadIdx.x;
  if (row >= rows) {                      // zero-fill padding rows
    dst[(size_t)row * DIM + t] = (_Float16)0.0f;
    return;
  }
  float x = src[(size_t)row * DIM + t];
  __shared__ float red[256];
  red[t] = x * x;
  __syncthreads();
  #pragma unroll
  for (int off = 128; off > 0; off >>= 1) {
    if (t < off) red[t] += red[t + off];
    __syncthreads();
  }
  float inv = 1.0f / fmaxf(sqrtf(red[0]), 1e-12f);
  dst[(size_t)row * DIM + t] = (_Float16)(x * inv);
}

// ---------------------------------------------------------------------------
// Kernel 2: sims = qn * dbn^T via v_wmma_f32_16x16x32_f16.
// Block: 256 thr (8 waves). Block tile: 128 q x 64 n (q + db staged in LDS,
// 96 KB). Wave w: rows [16w,16w+16), four 16x16 accumulators across n.
// K = 256 in 8 steps of 32, double-buffered from LDS. No bounds checks.
// ---------------------------------------------------------------------------
union AFrag { v16h v; v8h h[2]; };

__global__ void k_sims_wmma(const _Float16* __restrict__ qn,
                            const _Float16* __restrict__ dbn,
                            float* __restrict__ sims) {
  __shared__ __align__(32) _Float16 ldsQ[128 * DIM];  // 64 KB
  __shared__ __align__(32) _Float16 ldsB[64 * DIM];   // 32 KB

  const int t    = threadIdx.x;
  const int lane = t & 31;
  const int wave = t >> 5;
  const int q0   = blockIdx.x * 128;
  const int n0   = blockIdx.y * 64;

  // cooperative staging: q tile 2048 v16h, db tile 1024 v16h
  {
    const v16h* qg = (const v16h*)(qn + (size_t)q0 * DIM);
    v16h*       ql = (v16h*)ldsQ;
    #pragma unroll
    for (int i = 0; i < 8; ++i) ql[t + 256 * i] = qg[t + 256 * i];
    const v16h* bg = (const v16h*)(dbn + (size_t)n0 * DIM);
    v16h*       bl = (v16h*)ldsB;
    #pragma unroll
    for (int i = 0; i < 4; ++i) bl[t + 256 * i] = bg[t + 256 * i];
  }
  __syncthreads();

  const int row = lane & 15;
  const int hi  = lane >> 4;
  // A-frag lane base: wave's 16 q rows
  const _Float16* qbase = ldsQ + (wave * 16 + row) * DIM + hi * 16;
  // B-frag lane base: lane holds 16 contiguous K of db row (lane&15), K += 16*hi
  const _Float16* bbase = ldsB + row * DIM + hi * 16;

  v8f c0 = {}, c1 = {}, c2 = {}, c3 = {};

  // prologue: stage k-step 0 from LDS
  AFrag a; v16h b0, b1, b2, b3;
  a.h[0] = *(const v8h*)(qbase);
  a.h[1] = *(const v8h*)(qbase + 8);
  b0 = *(const v16h*)(bbase);
  b1 = *(const v16h*)(bbase + 16 * DIM);
  b2 = *(const v16h*)(bbase + 32 * DIM);
  b3 = *(const v16h*)(bbase + 48 * DIM);

  #pragma unroll
  for (int kk = 0; kk < 8; ++kk) {
    AFrag an; v16h bn0, bn1, bn2, bn3;
    if (kk < 7) {                          // prefetch next K-step from LDS
      const int k0 = (kk + 1) * 32;
      an.h[0] = *(const v8h*)(qbase + k0);
      an.h[1] = *(const v8h*)(qbase + k0 + 8);
      bn0 = *(const v16h*)(bbase + k0);
      bn1 = *(const v16h*)(bbase + k0 + 16 * DIM);
      bn2 = *(const v16h*)(bbase + k0 + 32 * DIM);
      bn3 = *(const v16h*)(bbase + k0 + 48 * DIM);
    }
    c0 = __builtin_amdgcn_wmma_f32_16x16x32_f16(false, a.v, false, b0,
                                                (short)0, c0, false, false);
    c1 = __builtin_amdgcn_wmma_f32_16x16x32_f16(false, a.v, false, b1,
                                                (short)0, c1, false, false);
    c2 = __builtin_amdgcn_wmma_f32_16x16x32_f16(false, a.v, false, b2,
                                                (short)0, c2, false, false);
    c3 = __builtin_amdgcn_wmma_f32_16x16x32_f16(false, a.v, false, b3,
                                                (short)0, c3, false, false);
    if (kk < 7) { a = an; b0 = bn0; b1 = bn1; b2 = bn2; b3 = bn3; }
  }

  // store: C/D layout VGPR r -> row r + 8*(lane>>4), col = lane&15
  const int mrow = q0 + wave * 16 + hi * 8;
  const int nl   = lane & 15;
  float* s0 = sims + (size_t)mrow * N_PAD + n0 + nl;
  #pragma unroll
  for (int r = 0; r < 8; ++r) {
    float* sr = s0 + (size_t)r * N_PAD;
    sr[ 0] = c0[r];
    sr[16] = c1[r];
    sr[32] = c2[r];
    sr[48] = c3[r];
  }
}

// ---------------------------------------------------------------------------
// Kernel 3: per-query top-70 (histogram threshold + candidate select),
// softmax(T=0.04), weighted gather of es_db. One 256-thread block per query.
// Only columns [0, N_DB) of each padded sims row are read.
// ---------------------------------------------------------------------------
#define HIST_BITS 13
#define NBINS     8192
#define CAND_CAP  2048

__device__ __forceinline__ unsigned sortable(float v) {
  unsigned u = __float_as_uint(v);
  return (u & 0x80000000u) ? ~u : (u | 0x80000000u);
}

__global__ void k_topk_softmax_gather(const float* __restrict__ sims,
                                      const float* __restrict__ es,
                                      float* __restrict__ out) {
  __shared__ unsigned hist[NBINS];      // 32 KB
  __shared__ unsigned partial[256];
  __shared__ float    cvals[CAND_CAP];  // 8 KB
  __shared__ int      cidx[CAND_CAP];   // 8 KB
  __shared__ unsigned ctr[2];           // [0]=cand count, [1]=bstar
  __shared__ float    rv[256];
  __shared__ int      ri[256];
  __shared__ float    topv[TOPK];
  __shared__ int      topi[TOPK];
  __shared__ float    wts[TOPK];

  const int q = blockIdx.x;
  const int t = threadIdx.x;
  const float* rowp = sims + (size_t)q * N_PAD;

  for (int i = t; i < NBINS; i += 256) hist[i] = 0u;
  if (t < 2) ctr[t] = 0u;
  __syncthreads();

  // pass 1: histogram of sortable-key top bits
  for (int i = t; i < N_DB; i += 256)
    atomicAdd(&hist[sortable(rowp[i]) >> (32 - HIST_BITS)], 1u);
  __syncthreads();

  // per-thread partial over 32 bins, then serial threshold search (top-down)
  unsigned p = 0;
  #pragma unroll
  for (int j = 0; j < 32; ++j) p += hist[t * 32 + j];
  partial[t] = p;
  __syncthreads();

  if (t == 0) {
    unsigned acc = 0; int tt;
    for (tt = 255; tt >= 0; --tt) {
      if (acc + partial[tt] >= TOPK) break;
      acc += partial[tt];
    }
    int bstar = 0;
    if (tt >= 0) {
      int b;
      for (b = tt * 32 + 31; b >= tt * 32; --b) {
        if (acc + hist[b] >= TOPK) break;
        acc += hist[b];
      }
      bstar = (b >= tt * 32) ? b : tt * 32;
    }
    ctr[1] = (unsigned)bstar;
  }
  __syncthreads();
  const unsigned bstar = ctr[1];

  // pass 2: collect candidates with bin >= bstar
  for (int i = t; i < N_DB; i += 256) {
    float v = rowp[i];
    if ((sortable(v) >> (32 - HIST_BITS)) >= bstar) {
      unsigned slot = atomicAdd(&ctr[0], 1u);
      if (slot < CAND_CAP) { cvals[slot] = v; cidx[slot] = i; }
    }
  }
  __syncthreads();
  const int m = (int)(ctr[0] < CAND_CAP ? ctr[0] : CAND_CAP);

  // 70 block-wide argmax extractions (set is permutation-invariant downstream)
  for (int k = 0; k < TOPK; ++k) {
    float lv = -1e30f; int li = -1;
    for (int i = t; i < m; i += 256)
      if (cvals[i] > lv) { lv = cvals[i]; li = i; }
    rv[t] = lv; ri[t] = li;
    __syncthreads();
    #pragma unroll
    for (int off = 128; off > 0; off >>= 1) {
      if (t < off && rv[t + off] > rv[t]) { rv[t] = rv[t + off]; ri[t] = ri[t + off]; }
      __syncthreads();
    }
    if (t == 0) {
      int w = ri[0];
      topv[k] = rv[0];
      topi[k] = (w >= 0) ? cidx[w] : 0;
      if (w >= 0) cvals[w] = -1e30f;
    }
    __syncthreads();
  }

  // softmax over the 70 (topv[0] is the max — extraction is descending)
  if (t == 0) {
    float mx = topv[0], s = 0.f;
    for (int k = 0; k < TOPK; ++k) {
      float e = expf((topv[k] - mx) * INV_TEMP);
      wts[k] = e; s += e;
    }
    float inv = 1.0f / s;
    for (int k = 0; k < TOPK; ++k) wts[k] *= inv;
  }
  __syncthreads();

  // weighted gather: thread t owns output dim t; per-k reads are coalesced
  float acc = 0.f;
  for (int k = 0; k < TOPK; ++k)
    acc += wts[k] * es[(size_t)topi[k] * DIM + t];
  out[(size_t)q * DIM + t] = acc;
}

// ---------------------------------------------------------------------------
extern "C" void kernel_launch(void* const* d_in, const int* in_sizes, int n_in,
                              void* d_out, int out_size, void* d_ws, size_t ws_size,
                              hipStream_t stream) {
  const float* query = (const float*)d_in[0];  // [1024, 256]
  const float* en_db = (const float*)d_in[1];  // [100000, 256]
  const float* es_db = (const float*)d_in[2];  // [100000, 256]
  float* out = (float*)d_out;                  // [1024, 256]

  char* ws = (char*)d_ws;
  const size_t qn_off   = 0;
  const size_t qn_bytes = (size_t)B_Q * DIM * sizeof(_Float16);      // 512 KB
  const size_t db_off   = qn_off + qn_bytes;
  const size_t db_bytes = (size_t)N_PAD * DIM * sizeof(_Float16);    // 51.2 MB
  const size_t sm_off   = db_off + db_bytes;                         // 256B aligned

  _Float16* qn   = (_Float16*)(ws + qn_off);
  _Float16* dbn  = (_Float16*)(ws + db_off);
  float*    sims = (float*)(ws + sm_off);                            // 410 MB

  k_normalize_f16<<<B_Q,   256, 0, stream>>>(query, qn, B_Q);
  k_normalize_f16<<<N_PAD, 256, 0, stream>>>(en_db, dbn, N_DB);

  dim3 g2(B_Q / 128, N_PAD / 64);   // 8 x 1563, exact cover, no guards
  k_sims_wmma<<<g2, 256, 0, stream>>>(qn, dbn, sims);

  k_topk_softmax_gather<<<B_Q, 256, 0, stream>>>(sims, es_db, out);
}
